// MDViT_DSN_57337813401611
// MI455X (gfx1250) — matrix operand order repaired and checked by reading, not compiled, not run on hardware
//
#include <hip/hip_runtime.h>
#include <hip/hip_bf16.h>
#include <math.h>

// ---------------------------------------------------------------------------
// MDViT block for MI455X (gfx1250, wave32, WMMA).
// All dense contractions use v_wmma_f32_16x16x32_bf16 (bf16 in, f32 acc).
// Fragments fed from LDS with ds_load_b128; each wave runs 4 independent
// accumulator tiles sharing one A fragment so WMMA hazard slots are filled
// with useful WMMAs instead of v_nops.
// ---------------------------------------------------------------------------

#define B_  8
#define CIN 64
#define HIN 128
#define C_  128
#define H_  64
#define W_  64
#define N_  4096           // H_*W_
#define T_  32768          // B_*N_
#define NH  8
#define CH  16             // C_/NH
#define HID 1024           // mlp hidden
#define LDBT 136           // padded [N][K] row stride (halves); 136*2=272=17*16

typedef __attribute__((ext_vector_type(16))) __bf16          v16bf;
typedef __attribute__((ext_vector_type(16))) unsigned short  v16u;
typedef __attribute__((ext_vector_type(8)))  unsigned short  v8u;
typedef __attribute__((ext_vector_type(8)))  float           v8f;

__device__ __forceinline__ unsigned short f2bf(float f) {
  unsigned int u = __builtin_bit_cast(unsigned int, f);
  unsigned int r = (u + 0x7FFFu + ((u >> 16) & 1u)) >> 16;   // RNE
  return (unsigned short)r;
}
__device__ __forceinline__ float bf2f(unsigned short h) {
  unsigned int u = ((unsigned int)h) << 16;
  return __builtin_bit_cast(float, u);
}

__device__ __forceinline__ v8f wmma_bf16(v16u a, v16u b, v8f c) {
  return __builtin_amdgcn_wmma_f32_16x16x32_bf16(
      false, __builtin_bit_cast(v16bf, a),
      false, __builtin_bit_cast(v16bf, b),
      (short)0, c, false, false);
}

// A fragment 16x32 bf16 (row-major in LDS, row stride lda halves, lda%8==0).
// ISA 7.12.2: lanes 0-15 row M=lane, K {0..7,16..23}; lanes 16-31 same rows,
// K {8..15,24..31}. Both runs are 16B-contiguous -> 2x ds_load_b128.
__device__ __forceinline__ v16u load_frag_a(const unsigned short* A, int lda,
                                            int lane) {
  int r  = lane & 15;
  int kb = (lane >> 4) << 3;           // 0 or 8
  const unsigned short* p = A + r * lda + kb;
  v8u lo = *(const v8u*)p;             // K = kb .. kb+7
  v8u hi = *(const v8u*)(p + 16);      // K = kb+16 .. kb+23
  return __builtin_shufflevector(lo, hi, 0, 1, 2, 3, 4, 5, 6, 7,
                                 8, 9, 10, 11, 12, 13, 14, 15);
}

// B fragment 32x16 bf16 from TRANSPOSED LDS tile Bt[n][k] (row stride ldbT
// halves, ldbT%8==0). Lane needs column n, K {kt..kt+7, kt+16..kt+23}
// (kt = 0 or 8, same as A) -> 2x ds_load_b128.
__device__ __forceinline__ v16u load_frag_bT(const unsigned short* Bt,
                                             int ldbT, int lane) {
  int n  = lane & 15;
  int kt = (lane >> 4) << 3;           // 0 or 8
  const unsigned short* p = Bt + n * ldbT + kt;
  v8u lo = *(const v8u*)p;
  v8u hi = *(const v8u*)(p + 16);
  return __builtin_shufflevector(lo, hi, 0, 1, 2, 3, 4, 5, 6, 7,
                                 8, 9, 10, 11, 12, 13, 14, 15);
}

// One wave accumulates FOUR adjacent 16x16 tiles (same 16 rows, 4 n-tiles)
// sharing a single A fragment per k-step. The 4 WMMAs per step are
// independent -> no v_nop hazard padding, and A LDS traffic is 1/4.
__device__ __forceinline__ void wave_gemm4(const unsigned short* A, int lda,
                                           const unsigned short* Bt, int ldbT,
                                           int ksteps, int lane, v8f acc[4]) {
  for (int ks = 0; ks < ksteps; ++ks) {
    v16u a  = load_frag_a(A + ks * 32, lda, lane);
    v16u b0 = load_frag_bT(Bt + 0 * 16 * ldbT + ks * 32, ldbT, lane);
    v16u b1 = load_frag_bT(Bt + 1 * 16 * ldbT + ks * 32, ldbT, lane);
    v16u b2 = load_frag_bT(Bt + 2 * 16 * ldbT + ks * 32, ldbT, lane);
    v16u b3 = load_frag_bT(Bt + 3 * 16 * ldbT + ks * 32, ldbT, lane);
    acc[0] = wmma_bf16(a, b0, acc[0]);
    acc[1] = wmma_bf16(a, b1, acc[1]);
    acc[2] = wmma_bf16(a, b2, acc[2]);
    acc[3] = wmma_bf16(a, b3, acc[3]);
  }
}

#define ZERO8 {0.f, 0.f, 0.f, 0.f, 0.f, 0.f, 0.f, 0.f}

// ---------------------------------------------------------------------------
// K0: domain gate (Linear->ReLU->Linear, softmax over heads) + per-domain
//     BN/LN parameter gather. 1 block x 128 threads.
//     prm layout: [0)bnscale [128)bnshift [256)ln1g [384)ln1b [512)ln2g [640)ln2b
// ---------------------------------------------------------------------------
__global__ void k0_prep(const float* __restrict__ lbl,
                        const float* __restrict__ w1, const float* __restrict__ b1,
                        const float* __restrict__ w2, const float* __restrict__ b2,
                        const float* __restrict__ bn_g, const float* __restrict__ bn_b,
                        const float* __restrict__ bn_m, const float* __restrict__ bn_v,
                        const float* __restrict__ n1g, const float* __restrict__ n1b,
                        const float* __restrict__ n2g, const float* __restrict__ n2b,
                        const int* __restrict__ dptr,
                        float* __restrict__ prm, float* __restrict__ da) {
  __shared__ float hid[64];
  __shared__ float pre[128];
  int tid = threadIdx.x;
  int d = dptr[0];
  {
    float sc = bn_g[d * 128 + tid] * rsqrtf(bn_v[d * 128 + tid] + 1e-5f);
    prm[tid]       = sc;
    prm[128 + tid] = bn_b[d * 128 + tid] - bn_m[d * 128 + tid] * sc;
    prm[256 + tid] = n1g[d * 128 + tid];
    prm[384 + tid] = n1b[d * 128 + tid];
    prm[512 + tid] = n2g[d * 128 + tid];
    prm[640 + tid] = n2b[d * 128 + tid];
  }
  for (int b = 0; b < B_; ++b) {
    __syncthreads();
    if (tid < 64) {
      float s = b1[tid];
      for (int i = 0; i < 4; ++i) s += lbl[b * 4 + i] * w1[i * 64 + tid];
      hid[tid] = fmaxf(s, 0.f);
    }
    __syncthreads();
    {
      float s = b2[tid];
      for (int j = 0; j < 64; ++j) s += hid[j] * w2[j * 128 + tid];
      pre[tid] = s;
    }
    __syncthreads();
    {
      int ch = tid & 15;
      float m = -3.4e38f;
      for (int hh = 0; hh < NH; ++hh) m = fmaxf(m, pre[hh * 16 + ch]);
      float ssum = 0.f;
      for (int hh = 0; hh < NH; ++hh) ssum += __expf(pre[hh * 16 + ch] - m);
      da[b * 128 + tid] = __expf(pre[tid] - m) / ssum;
    }
  }
}

// ---------------------------------------------------------------------------
// K1: depthwise 3x3 stride-2 conv (SAME: pad_lo=0, pad_hi=1) -> bf16 tokens
//     xdw[t][cin], t = b*4096 + oy*64 + ox.
// ---------------------------------------------------------------------------
__global__ void k1_dwconv(const float* __restrict__ img,
                          const float* __restrict__ wdw,
                          unsigned short* __restrict__ xdw) {
  int idx = blockIdx.x * blockDim.x + threadIdx.x;
  int cin = idx & 63;
  int t   = idx >> 6;
  int b = t >> 12, oy = (t >> 6) & 63, ox = t & 63;
  const float* ip = img + ((size_t)b * CIN + cin) * (HIN * HIN);
  const float* wp = wdw + cin * 9;
  float acc = 0.f;
#pragma unroll
  for (int ky = 0; ky < 3; ++ky) {
    int iy = 2 * oy + ky;
    if (iy >= HIN) continue;
#pragma unroll
    for (int kx = 0; kx < 3; ++kx) {
      int ix = 2 * ox + kx;
      if (ix >= HIN) continue;
      acc += ip[iy * HIN + ix] * wp[ky * 3 + kx];
    }
  }
  xdw[(size_t)t * 64 + cin] = f2bf(acc);
}

// ---------------------------------------------------------------------------
// K2: pointwise conv as WMMA GEMM [T x 64] @ [64 x 128] + BN + hardswish.
//     pe_pw_w is already [N][K] -> direct bf16 copy is the transposed tile.
// ---------------------------------------------------------------------------
__global__ void k2_pointwise(const unsigned short* __restrict__ xdw,
                             const float* __restrict__ pw,     // [128][64]
                             const float* __restrict__ prm,
                             float* __restrict__ x0) {
  extern __shared__ __attribute__((aligned(16))) char smem[];
  unsigned short* As = (unsigned short*)smem;    // 128x64
  unsigned short* Bt = As + 128 * 64;            // [n=128][k=64]
  int tid = threadIdx.x;
  size_t row0 = (size_t)blockIdx.x * 128;
  for (int i = tid * 8; i < 128 * 64; i += 256 * 8)
    *(v8u*)(As + i) = *(const v8u*)(xdw + row0 * 64 + i);   // 16B copies
  for (int i = tid; i < 128 * 64; i += 256)
    Bt[i] = f2bf(pw[i]);                                    // already [N][K]
  __syncthreads();
  int lane = tid & 31, wave = tid >> 5, m0 = wave * 16;
  for (int g = 0; g < 2; ++g) {
    v8f acc[4] = {ZERO8, ZERO8, ZERO8, ZERO8};
    wave_gemm4(As + m0 * 64, 64, Bt + g * 4 * 16 * 64, 64, 2, lane, acc);
#pragma unroll
    for (int j = 0; j < 4; ++j) {
      int col = (g * 4 + j) * 16 + (lane & 15);
      float sc = prm[col], sh = prm[128 + col];
#pragma unroll
      for (int i = 0; i < 8; ++i) {
        int row = m0 + ((lane >> 4) << 3) + i;
        float v = acc[j][i] * sc + sh;
        v = v * fminf(fmaxf(v + 3.f, 0.f), 6.f) * (1.f / 6.f); // hardswish
        x0[(row0 + row) * 128 + col] = v;
      }
    }
  }
}

// ---------------------------------------------------------------------------
// K3: ConvPosEnc dw3x3 (pad 1) + residual. Channel-last layout is coalesced.
// ---------------------------------------------------------------------------
__global__ void k3_cpe(const float* __restrict__ x0,
                       const float* __restrict__ w, const float* __restrict__ bias,
                       float* __restrict__ x) {
  int idx = blockIdx.x * blockDim.x + threadIdx.x;
  int c = idx & 127;
  int t = idx >> 7;
  int b = t >> 12, oy = (t >> 6) & 63, ox = t & 63;
  const float* wp = w + c * 9;
  float acc = bias[c];
#pragma unroll
  for (int ky = 0; ky < 3; ++ky) {
    int iy = oy + ky - 1;
    if ((unsigned)iy >= (unsigned)H_) continue;
#pragma unroll
    for (int kx = 0; kx < 3; ++kx) {
      int ix = ox + kx - 1;
      if ((unsigned)ix >= (unsigned)W_) continue;
      acc += x0[((size_t)b * N_ + iy * W_ + ix) * 128 + c] * wp[ky * 3 + kx];
    }
  }
  x[(size_t)t * 128 + c] = x0[(size_t)t * 128 + c] + acc;
}

// ---------------------------------------------------------------------------
// K4: LN1 + QKV GEMM [T x 128] @ [128 x 384], chunked N=128 (q,k,v).
// ---------------------------------------------------------------------------
__global__ void k4_ln_qkv(const float* __restrict__ x,
                          const float* __restrict__ qkv_w,
                          const float* __restrict__ qkv_b,
                          const float* __restrict__ prm,
                          unsigned short* __restrict__ qkv) {
  extern __shared__ __attribute__((aligned(16))) char smem[];
  unsigned short* As = (unsigned short*)smem;     // 128x128
  unsigned short* Bt = As + 128 * 128;            // [n=128][k=128] stride LDBT
  int tid = threadIdx.x;
  size_t row0 = (size_t)blockIdx.x * 128;
  if (tid < 128) {
    const float* xr = x + (row0 + tid) * 128;
    __builtin_prefetch(xr, 0, 0);                 // global_prefetch_b8
    float m = 0.f;
    for (int c = 0; c < 128; ++c) m += xr[c];
    m *= (1.f / 128.f);
    float v = 0.f;
    for (int c = 0; c < 128; ++c) { float d2 = xr[c] - m; v += d2 * d2; }
    v *= (1.f / 128.f);
    float inv = rsqrtf(v + 1e-6f);
    for (int c = 0; c < 128; ++c)
      As[tid * 128 + c] = f2bf((xr[c] - m) * inv * prm[256 + c] + prm[384 + c]);
  }
  int lane = tid & 31, wave = tid >> 5, m0 = wave * 16;
  for (int s = 0; s < 3; ++s) {
    __syncthreads();
    for (int i = tid; i < 128 * 128; i += 256) {
      int k = i >> 7, n = i & 127;                // coalesced global read
      Bt[n * LDBT + k] = f2bf(qkv_w[k * 384 + s * 128 + n]);
    }
    __syncthreads();
    for (int g = 0; g < 2; ++g) {
      v8f acc[4] = {ZERO8, ZERO8, ZERO8, ZERO8};
      wave_gemm4(As + m0 * 128, 128, Bt + g * 4 * 16 * LDBT, LDBT, 4, lane, acc);
#pragma unroll
      for (int j = 0; j < 4; ++j) {
        int col = (g * 4 + j) * 16 + (lane & 15);
        float bb = qkv_b[s * 128 + col];
#pragma unroll
        for (int i = 0; i < 8; ++i) {
          int row = m0 + ((lane >> 4) << 3) + i;
          qkv[(row0 + row) * 384 + s * 128 + col] = f2bf(acc[j][i] + bb);
        }
      }
    }
  }
}

// ---------------------------------------------------------------------------
// K5: per-(b,h,ch) column max of K over the 4096 tokens (softmax stability).
// ---------------------------------------------------------------------------
__global__ void k5_kmax(const unsigned short* __restrict__ qkv,
                        float* __restrict__ kmax) {
  __shared__ float red[256];
  int cid = blockIdx.x;                  // b*128 + (h*16+ch)
  int b = cid >> 7, c = cid & 127;
  const unsigned short* kp = qkv + (size_t)b * N_ * 384 + 128 + c;
  float m = -3.4e38f;
  for (int n = threadIdx.x; n < N_; n += 256)
    m = fmaxf(m, bf2f(kp[(size_t)n * 384]));
  red[threadIdx.x] = m;
  __syncthreads();
  for (int s = 128; s > 0; s >>= 1) {
    if (threadIdx.x < (unsigned)s)
      red[threadIdx.x] = fmaxf(red[threadIdx.x], red[threadIdx.x + s]);
    __syncthreads();
  }
  if (threadIdx.x == 0) kmax[cid] = red[0];
}

// ---------------------------------------------------------------------------
// K6: partial KV = E^T V over 256-token segments via WMMA, plus partial
//     exp-sums. block = (b,h,seg). V staged transposed ([v][n]).
// ---------------------------------------------------------------------------
__global__ void k6_kv_partial(const unsigned short* __restrict__ qkv,
                              const float* __restrict__ kmax,
                              float* __restrict__ kvpart,
                              float* __restrict__ esump) {
  __shared__ __attribute__((aligned(16))) unsigned short E[16 * 256];  // [ch][n]
  __shared__ __attribute__((aligned(16))) unsigned short Vt[16 * 256]; // [v][n]
  __shared__ float kvtile[256];
  int tid = threadIdx.x;
  int bh = blockIdx.x >> 4, seg = blockIdx.x & 15;
  int b = bh >> 3, hh = bh & 7;
  size_t tok = (size_t)b * N_ + seg * 256 + tid;
  const unsigned short* base = qkv + tok * 384;
#pragma unroll
  for (int ch = 0; ch < 16; ++ch) {
    float e = __expf(bf2f(base[128 + hh * 16 + ch]) - kmax[b * 128 + hh * 16 + ch]);
    E[ch * 256 + tid] = f2bf(e);
  }
#pragma unroll
  for (int v = 0; v < 16; ++v) Vt[v * 256 + tid] = base[256 + hh * 16 + v];
  kvtile[tid] = 0.f;
  __syncthreads();
  int lane = tid & 31, wave = tid >> 5;
  {
    v16u a  = load_frag_a(E + wave * 32, 256, lane);        // E 16 x 32 chunk
    v16u bb = load_frag_bT(Vt + wave * 32, 256, lane);      // V 32 x 16 chunk
    v8f acc = ZERO8;
    acc = wmma_bf16(a, bb, acc);
    int col = lane & 15;
#pragma unroll
    for (int i = 0; i < 8; ++i) {
      int row = ((lane >> 4) << 3) + i;
      atomicAdd(&kvtile[row * 16 + col], acc[i]);           // LDS reduce 8 waves
    }
  }
  __syncthreads();
  kvpart[(size_t)blockIdx.x * 256 + tid] = kvtile[tid];
  if (tid < 16) {
    float s = 0.f;
    for (int n = 0; n < 256; ++n) s += bf2f(E[tid * 256 + n]);
    esump[blockIdx.x * 16 + tid] = s;
  }
}

// ---------------------------------------------------------------------------
// K7: reduce 16 segments, row-normalize by exp-sum, fold in Ch^-0.5 = 0.25.
// ---------------------------------------------------------------------------
__global__ void k7_kv_norm(const float* __restrict__ kvpart,
                           const float* __restrict__ esump,
                           float* __restrict__ kvn) {
  int bh = blockIdx.x, tid = threadIdx.x;      // tid = k*16+v
  int krow = tid >> 4;
  float s = 0.f, es = 0.f;
  for (int seg = 0; seg < 16; ++seg) {
    s  += kvpart[((size_t)bh * 16 + seg) * 256 + tid];
    es += esump[(bh * 16 + seg) * 16 + krow];
  }
  kvn[bh * 256 + tid] = 0.25f * s / es;
}

// ---------------------------------------------------------------------------
// K8: CRPE depthwise convs on V image: ch<32 ->3x3, <80 ->5x5, else 7x7.
// ---------------------------------------------------------------------------
__global__ void k8_crpe(const unsigned short* __restrict__ qkv,
                        const float* __restrict__ w3, const float* __restrict__ b3,
                        const float* __restrict__ w5, const float* __restrict__ b5,
                        const float* __restrict__ w7, const float* __restrict__ b7,
                        unsigned short* __restrict__ cv) {
  int idx = blockIdx.x * blockDim.x + threadIdx.x;
  int c = idx & 127;
  int t = idx >> 7;
  int b = t >> 12, oy = (t >> 6) & 63, ox = t & 63;
  const float* wp;
  float acc;
  int ksz;
  if (c < 32)      { ksz = 3; wp = w3 + c * 9;          acc = b3[c]; }
  else if (c < 80) { ksz = 5; wp = w5 + (c - 32) * 25;  acc = b5[c - 32]; }
  else             { ksz = 7; wp = w7 + (c - 80) * 49;  acc = b7[c - 80]; }
  int pad = ksz >> 1;
  const unsigned short* vb = qkv + 256 + c;
  size_t bbase = (size_t)b * N_;
  for (int ky = 0; ky < ksz; ++ky) {
    int iy = oy + ky - pad;
    if ((unsigned)iy >= (unsigned)H_) continue;
    for (int kx = 0; kx < ksz; ++kx) {
      int ix = ox + kx - pad;
      if ((unsigned)ix >= (unsigned)W_) continue;
      acc += bf2f(vb[(bbase + iy * W_ + ix) * 384]) * wp[ky * ksz + kx];
    }
  }
  cv[(size_t)t * 128 + c] = f2bf(acc);
}

// ---------------------------------------------------------------------------
// K9: att = da * (q @ kvn + q*cv); x2 = x + att @ proj_w + proj_b (WMMA).
// ---------------------------------------------------------------------------
__global__ void k9_att_proj(const unsigned short* __restrict__ qkv,
                            const float* __restrict__ kvn,
                            const unsigned short* __restrict__ cv,
                            const float* __restrict__ da,
                            const float* __restrict__ proj_w,
                            const float* __restrict__ proj_b,
                            const float* __restrict__ x,
                            float* __restrict__ x2) {
  extern __shared__ __attribute__((aligned(16))) char smem[];
  float* kvs = (float*)smem;                               // 2048 f32
  float* das = kvs + 2048;                                 // 128 f32
  unsigned short* att = (unsigned short*)(das + 128);      // 128x128 bf16
  unsigned short* Bt  = att + 128 * 128;                   // [128][LDBT]
  int tid = threadIdx.x;
  size_t row0 = (size_t)blockIdx.x * 128;
  int b = blockIdx.x >> 5;                                 // 32 blocks per image
  for (int i = tid; i < 2048; i += 256) kvs[i] = kvn[b * 2048 + i];
  if (tid < 128) das[tid] = da[b * 128 + tid];
  for (int i = tid; i < 128 * 128; i += 256) {
    int k = i >> 7, n = i & 127;
    Bt[n * LDBT + k] = f2bf(proj_w[i]);                    // proj_w[k][n]
  }
  __syncthreads();
  for (int i = tid; i < 128 * 128; i += 256) {
    int rr = i >> 7, cc = i & 127;
    int hh = cc >> 4, chv = cc & 15;
    const unsigned short* qrow = qkv + (row0 + rr) * 384;  // q = cols [0,128)
    float s = 0.f;
#pragma unroll
    for (int k2 = 0; k2 < 16; ++k2)
      s += bf2f(qrow[hh * 16 + k2]) * kvs[(hh * 16 + k2) * 16 + chv];
    s += bf2f(qrow[cc]) * bf2f(cv[(row0 + rr) * 128 + cc]);  // crpe (unscaled)
    att[i] = f2bf(das[cc] * s);
  }
  __syncthreads();
  int lane = tid & 31, wave = tid >> 5, m0 = wave * 16;
  for (int g = 0; g < 2; ++g) {
    v8f acc[4] = {ZERO8, ZERO8, ZERO8, ZERO8};
    wave_gemm4(att + m0 * 128, 128, Bt + g * 4 * 16 * LDBT, LDBT, 4, lane, acc);
#pragma unroll
    for (int j = 0; j < 4; ++j) {
      int col = (g * 4 + j) * 16 + (lane & 15);
      float bb = proj_b[col];
#pragma unroll
      for (int i = 0; i < 8; ++i) {
        int row = m0 + ((lane >> 4) << 3) + i;
        size_t o = (row0 + row) * 128 + col;
        x2[o] = x[o] + acc[j][i] + bb;
      }
    }
  }
}

// ---------------------------------------------------------------------------
// K10: fused MLP: LN2 -> fc1(WMMA) -> exact GELU -> fc2(WMMA) -> +residual.
//     64-token tile; hidden tile (64x1024 bf16) lives entirely in LDS
//     (within the 320KB WGP LDS). Each wave owns a 16x64 output strip:
//     rg = wave>>1 (rows), ntb = (wave&1)*4 (4 n-tiles).
// ---------------------------------------------------------------------------
__global__ void k10_mlp(const float* __restrict__ x2,
                        const float* __restrict__ prm,
                        const float* __restrict__ fc1_w, const float* __restrict__ fc1_b,
                        const float* __restrict__ fc2_w, const float* __restrict__ fc2_b,
                        float* __restrict__ out) {
  extern __shared__ __attribute__((aligned(16))) char smem[];
  unsigned short* As = (unsigned short*)smem;   // 64x128
  unsigned short* Hs = As + 64 * 128;           // 64x1024
  unsigned short* Bt = Hs + 64 * 1024;          // [128][LDBT]
  int tid = threadIdx.x;
  size_t row0 = (size_t)blockIdx.x * 64;
  if (tid < 64) {
    const float* xr = x2 + (row0 + tid) * 128;
    __builtin_prefetch(xr, 0, 0);
    float m = 0.f;
    for (int c = 0; c < 128; ++c) m += xr[c];
    m *= (1.f / 128.f);
    float v = 0.f;
    for (int c = 0; c < 128; ++c) { float d2 = xr[c] - m; v += d2 * d2; }
    v *= (1.f / 128.f);
    float inv = rsqrtf(v + 1e-6f);
    for (int c = 0; c < 128; ++c)
      As[tid * 128 + c] = f2bf((xr[c] - m) * inv * prm[512 + c] + prm[640 + c]);
  }
  int lane = tid & 31, wave = tid >> 5;
  int rg = wave >> 1, ntb = (wave & 1) * 4, m0 = rg * 16;
  // ---- fc1 + GELU, 8 column chunks of 128 ----
  for (int nc = 0; nc < 8; ++nc) {
    __syncthreads();
    for (int i = tid; i < 128 * 128; i += 256) {
      int k = i >> 7, n = i & 127;
      Bt[n * LDBT + k] = f2bf(fc1_w[k * HID + nc * 128 + n]);
    }
    __syncthreads();
    v8f acc[4] = {ZERO8, ZERO8, ZERO8, ZERO8};
    wave_gemm4(As + m0 * 128, 128, Bt + ntb * 16 * LDBT, LDBT, 4, lane, acc);
#pragma unroll
    for (int j = 0; j < 4; ++j) {
      int colL = (ntb + j) * 16 + (lane & 15);
      float bb = fc1_b[nc * 128 + colL];
#pragma unroll
      for (int i = 0; i < 8; ++i) {
        int row = m0 + ((lane >> 4) << 3) + i;
        float g = acc[j][i] + bb;
        g = 0.5f * g * (1.f + erff(g * 0.70710678118654752f)); // exact GELU
        Hs[row * HID + nc * 128 + colL] = f2bf(g);
      }
    }
  }
  // ---- fc2, K=1024 in 8 blocks of 128; 4 accumulators persist across kb ----
  v8f acc2[4] = {ZERO8, ZERO8, ZERO8, ZERO8};
  for (int kb = 0; kb < 8; ++kb) {
    __syncthreads();
    for (int i = tid; i < 128 * 128; i += 256) {
      int k = i >> 7, n = i & 127;
      Bt[n * LDBT + k] = f2bf(fc2_w[(kb * 128 + k) * 128 + n]);
    }
    __syncthreads();
    wave_gemm4(Hs + m0 * HID + kb * 128, HID, Bt + ntb * 16 * LDBT, LDBT, 4,
               lane, acc2);
  }
#pragma unroll
  for (int j = 0; j < 4; ++j) {
    int col = (ntb + j) * 16 + (lane & 15);
    float bb = fc2_b[col];
#pragma unroll
    for (int i = 0; i < 8; ++i) {
      int row = m0 + ((lane >> 4) << 3) + i;
      size_t o = (row0 + row) * 128 + col;
      out[o] = x2[o] + acc2[j][i] + bb;
    }
  }
}

// ---------------------------------------------------------------------------
extern "C" void kernel_launch(void* const* d_in, const int* in_sizes, int n_in,
                              void* d_out, int out_size, void* d_ws, size_t ws_size,
                              hipStream_t stream) {
  const float* img     = (const float*)d_in[0];
  const float* lbl     = (const float*)d_in[1];
  const float* pe_dw_w = (const float*)d_in[2];
  const float* pe_pw_w = (const float*)d_in[3];
  const float* pe_bn_g = (const float*)d_in[4];
  const float* pe_bn_b = (const float*)d_in[5];
  const float* pe_bn_m = (const float*)d_in[6];
  const float* pe_bn_v = (const float*)d_in[7];
  const float* cpe_w   = (const float*)d_in[8];
  const float* cpe_b   = (const float*)d_in[9];
  const float* n1g     = (const float*)d_in[10];
  const float* n1b     = (const float*)d_in[11];
  const float* qkv_w   = (const float*)d_in[12];
  const float* qkv_b   = (const float*)d_in[13];
  const float* w3      = (const float*)d_in[14];
  const float* b3      = (const float*)d_in[15];
  const float* w5      = (const float*)d_in[16];
  const float* b5      = (const float*)d_in[17];
  const float* w7      = (const float*)d_in[18];
  const float* b7      = (const float*)d_in[19];
  const float* dom_w1  = (const float*)d_in[20];
  const float* dom_b1  = (const float*)d_in[21];
  const float* dom_w2  = (const float*)d_in[22];
  const float* dom_b2  = (const float*)d_in[23];
  const float* proj_w  = (const float*)d_in[24];
  const float* proj_b  = (const float*)d_in[25];
  const float* n2g     = (const float*)d_in[26];
  const float* n2b     = (const float*)d_in[27];
  const float* fc1_w   = (const float*)d_in[28];
  const float* fc1_b   = (const float*)d_in[29];
  const float* fc2_w   = (const float*)d_in[30];
  const float* fc2_b   = (const float*)d_in[31];
  const int*   dptr    = (const int*)d_in[32];
  (void)in_sizes; (void)n_in; (void)out_size; (void)ws_size;

  char* ws = (char*)d_ws;
  size_t off = 0;
  auto carve = [&](size_t bytes) -> char* {
    char* p = ws + off;
    off = (off + bytes + 255) & ~(size_t)255;
    return p;
  };
  float*          prm    = (float*)carve(768 * 4);
  float*          da     = (float*)carve(B_ * 128 * 4);
  unsigned short* xdw    = (unsigned short*)carve((size_t)T_ * 64 * 2);
  float*          x0     = (float*)carve((size_t)T_ * 128 * 4);
  float*          x      = (float*)carve((size_t)T_ * 128 * 4);
  unsigned short* qkv    = (unsigned short*)carve((size_t)T_ * 384 * 2);
  float*          kmax   = (float*)carve(1024 * 4);
  float*          kvpart = (float*)carve((size_t)64 * 16 * 256 * 4);
  float*          esump  = (float*)carve((size_t)64 * 16 * 16 * 4);
  float*          kvn    = (float*)carve((size_t)64 * 256 * 4);
  unsigned short* cv     = (unsigned short*)carve((size_t)T_ * 128 * 2);
  float*          x2     = (float*)carve((size_t)T_ * 128 * 4);
  float*          out    = (float*)d_out;

  k0_prep<<<1, 128, 0, stream>>>(lbl, dom_w1, dom_b1, dom_w2, dom_b2,
                                 pe_bn_g, pe_bn_b, pe_bn_m, pe_bn_v,
                                 n1g, n1b, n2g, n2b, dptr, prm, da);
  k1_dwconv<<<(T_ * 64) / 256, 256, 0, stream>>>(img, pe_dw_w, xdw);
  k2_pointwise<<<T_ / 128, 256, (128 * 64 + 128 * 64) * 2, stream>>>(
      xdw, pe_pw_w, prm, x0);
  k3_cpe<<<(T_ * 128) / 256, 256, 0, stream>>>(x0, cpe_w, cpe_b, x);
  k4_ln_qkv<<<T_ / 128, 256, (128 * 128 + 128 * LDBT) * 2, stream>>>(
      x, qkv_w, qkv_b, prm, qkv);
  k5_kmax<<<1024, 256, 0, stream>>>(qkv, kmax);
  k6_kv_partial<<<1024, 256, 0, stream>>>(qkv, kmax, kvpart, esump);
  k7_kv_norm<<<64, 256, 0, stream>>>(kvpart, esump, kvn);
  k8_crpe<<<(T_ * 128) / 256, 256, 0, stream>>>(qkv, w3, b3, w5, b5, w7, b7, cv);
  k9_att_proj<<<T_ / 128, 256,
                2048 * 4 + 128 * 4 + (128 * 128 + 128 * LDBT) * 2, stream>>>(
      qkv, kvn, cv, da, proj_w, proj_b, x, x2);
  k10_mlp<<<T_ / 64, 256,
            (64 * 128 + 64 * HID + 128 * LDBT) * 2, stream>>>(
      x2, prm, fc1_w, fc1_b, fc2_w, fc2_b, out);
}
